// DynamicDWConv_my_14173392076924
// MI455X (gfx1250) — matrix operand order repaired and checked by reading, not compiled, not run on hardware
//
#include <hip/hip_runtime.h>
#include <hip/hip_bf16.h>
#include <math.h>

// ---------------------------------------------------------------------------
// DynamicDWConv for MI455X (gfx1250, wave32)
//   x:[32,96,128,128] f32.  Pipeline:
//   1) pool_kernel : global avg pool -> pooled[32,96]        (reads 201MB)
//   2) dynw_kernel : h1 = sigmoid(BN(pooled@w1^T)) [32,24]
//                    wdyn = h1@w2^T + b2  [32,864] via V_WMMA_F32_16X16X32_F16
//   3) dwconv_kernel: per-sample depthwise 3x3, pad 1        (reads+writes 402MB)
// Total HBM traffic ~604MB -> ~26us floor at 23.3 TB/s. Compute is negligible,
// so kernels are built for b128 data movement; WMMA handles the matmul stage.
// ---------------------------------------------------------------------------

#define B_  32
#define C_  96
#define CR_ 24
#define H_  128
#define W_  128
#define KK_ 9             // 3x3
#define NOUT_ (C_ * KK_)  // 864
#define HW_ (H_ * W_)

typedef __attribute__((ext_vector_type(16))) _Float16 v16h;
typedef __attribute__((ext_vector_type(8)))  float    v8f;
typedef __attribute__((ext_vector_type(4)))  float    vf4;

// ---------------------------------------------------------------------------
// Kernel 1: global average pool over 128x128 per (b,c).  One block per (b,c).
// b128 loads, wave32 shfl reduce, cross-wave LDS reduce.
// ---------------------------------------------------------------------------
__global__ __launch_bounds__(256) void pool_kernel(const float* __restrict__ x,
                                                   float* __restrict__ pooled) {
    const int bc = blockIdx.x;                       // 0..3071
    const vf4* p = (const vf4*)(x + (size_t)bc * HW_);
    float s = 0.f;
    for (int i = threadIdx.x; i < HW_ / 4; i += 256) {   // 4096 float4
        vf4 v = p[i];
        s += v.x + v.y + v.z + v.w;
    }
    #pragma unroll
    for (int off = 16; off > 0; off >>= 1)
        s += __shfl_down(s, off, 32);
    __shared__ float wsum[8];
    const int lane = threadIdx.x & 31, wv = threadIdx.x >> 5;
    if (lane == 0) wsum[wv] = s;
    __syncthreads();
    if (threadIdx.x == 0) {
        float t = 0.f;
        #pragma unroll
        for (int i = 0; i < 8; ++i) t += wsum[i];
        pooled[bc] = t * (1.f / (float)HW_);
    }
}

// ---------------------------------------------------------------------------
// Kernel 2: dynamic weights. Single 256-thread (8-wave) block.
//   Stage A (scalar, tiny): h1[32][24] = sigmoid(BN(pooled @ w1^T)) -> LDS f16,
//     zero-padded to K=32.  Also stage w2 -> LDS f16 [864][32], zero-padded.
//   Stage B (WMMA): wdyn[32][864] = h1 @ w2^T + b2.
//   M=32 -> 2 tiles, N=864 -> 54 tiles, K=24 padded to 32 -> 1 WMMA per tile.
//   All fragment fills are unconditional LDS reads (no exec-mask branches).
// ---------------------------------------------------------------------------
__global__ __launch_bounds__(256) void dynw_kernel(const float* __restrict__ pooled,
                                                   const float* __restrict__ w1,
                                                   const float* __restrict__ gamma,
                                                   const float* __restrict__ beta,
                                                   const float* __restrict__ rmean,
                                                   const float* __restrict__ rvar,
                                                   const float* __restrict__ w2,
                                                   const float* __restrict__ b2,
                                                   float* __restrict__ wdyn) {
    __shared__ _Float16 h1s[32][32];                 // [M][K], K 24..31 zero
    __shared__ _Float16 w2s[NOUT_][32];              // [N][K], K 24..31 zero
    const int t = threadIdx.x;

    // zero h1s fully and the K=24..31 pad columns of w2s (as dwords)
    for (int i = t; i < 32 * 32 / 2; i += 256)
        ((unsigned*)&h1s[0][0])[i] = 0u;
    for (int i = t; i < NOUT_ * 4; i += 256) {       // 4 dwords = 8 halfs per row
        const int n = i >> 2, j = 12 + (i & 3);      // dword 12..15 = halfs 24..31
        ((unsigned*)&w2s[0][0])[n * 16 + j] = 0u;
    }

    // stage w2 -> LDS f16: 864*24 f32 = 5184 float4 (b128 loads)
    const vf4* w2v = (const vf4*)w2;
    for (int i = t; i < NOUT_ * CR_ / 4; i += 256) {
        const int n = i / 6, k0 = (i % 6) * 4;
        vf4 v = w2v[i];
        w2s[n][k0 + 0] = (_Float16)v.x;
        w2s[n][k0 + 1] = (_Float16)v.y;
        w2s[n][k0 + 2] = (_Float16)v.z;
        w2s[n][k0 + 3] = (_Float16)v.w;
    }

    // h1: 32*24 = 768 dot products of length 96
    #pragma unroll
    for (int r = 0; r < 3; ++r) {
        const int idx = t + 256 * r;                 // 0..767
        if (idx < B_ * CR_) {
            const int m = idx / CR_, j = idx % CR_;
            float acc = 0.f;
            const float* pr = pooled + m * C_;
            const float* wr = w1 + j * C_;
            #pragma unroll 4
            for (int k = 0; k < C_; ++k) acc += pr[k] * wr[k];
            acc = (acc - rmean[j]) * rsqrtf(rvar[j] + 1e-5f) * gamma[j] + beta[j];
            acc = 1.f / (1.f + __expf(-acc));        // sigmoid
            h1s[m][j] = (_Float16)acc;
        }
    }
    __syncthreads();

    const int wave = t >> 5, lane = t & 31;
    const int lhi = lane >> 4;                       // lane group (0/1)
    const int lm  = lane & 15;                       // M row / N col within tile

    // 108 tiles over 8 waves
    for (int tile = wave; tile < 2 * (NOUT_ / 16); tile += 8) {
        const int mt = tile & 1;                     // 0..1  (M tile)
        const int nt = tile >> 1;                    // 0..53 (N tile)
        const int n  = nt * 16 + lm;

        // A fragment: 16x32 f16.  half h -> K = 16*lhi + h
        v16h a;
        #pragma unroll
        for (int h = 0; h < 16; ++h) {
            const int K = 16 * lhi + h;
            a[h] = h1s[16 * mt + lm][K];
        }
        // B fragment: 32x16 f16.  B[k][n] = w2[n][k] (wdyn = h1 @ w2^T)
        // half h -> K = 16*lhi + h : 16 contiguous halfs from LDS row n.
        v16h bfr;
        #pragma unroll
        for (int h = 0; h < 16; ++h)
            bfr[h] = w2s[n][16 * lhi + h];

        v8f c = {};
        c = __builtin_amdgcn_wmma_f32_16x16x32_f16(
                /*neg_a=*/false, a, /*neg_b=*/false, bfr,
                /*c_mod=*/(short)0, c, /*reuse_a=*/false, /*reuse_b=*/false);

        const float bb = b2[n];
        #pragma unroll
        for (int v = 0; v < 8; ++v) {                // D: M = 16*mt + v + 8*lhi
            const int m = 16 * mt + v + 8 * lhi;
            wdyn[m * NOUT_ + n] = c[v] + bb;
        }
    }
}

// ---------------------------------------------------------------------------
// Kernel 3: per-sample depthwise 3x3, stride 1, pad 1.
// One 256-thread block per (b,c) slice; 16 passes of 8 output rows.
// Per pass: 10 input rows (zero-padded) -> LDS via b128 loads; each thread
// computes a 4-wide output quad and writes one nontemporal b128 store.
// ---------------------------------------------------------------------------
__global__ __launch_bounds__(256) void dwconv_kernel(const float* __restrict__ x,
                                                     const float* __restrict__ wdyn,
                                                     const float* __restrict__ bias,
                                                     float* __restrict__ out) {
    const int tid = threadIdx.x;                     // 0..255
    const int bc  = blockIdx.x;                      // b*96 + c
    const int c   = bc % C_;

    __shared__ float lds[10][132];                   // [row][1+x], halos at 0 & 129

    const float* wp = wdyn + (size_t)bc * KK_;
    const float w00 = wp[0], w01 = wp[1], w02 = wp[2];
    const float w10 = wp[3], w11 = wp[4], w12 = wp[5];
    const float w20 = wp[6], w21 = wp[7], w22 = wp[8];
    const float bv = bias[c];

    const float* src = x   + (size_t)bc * HW_;
    float*       dst = out + (size_t)bc * HW_;

    if (tid < 10) {                                  // halo columns, written once
        lds[tid][0]      = 0.f;
        lds[tid][W_ + 1] = 0.f;
    }

    const int r  = tid >> 5;                         // output row in pass (0..7)
    const int q  = tid & 31;                         // quad index (x0 = 4q)
    const int x0 = q << 2;

    for (int pass = 0; pass < H_ / 8; ++pass) {
        const int ybase = pass * 8 - 1;              // lds row 0 == input row ybase
        __syncthreads();                             // prev compute done (+halo init)

        // load 10 rows x 32 quads = 320 b128 loads across 256 lanes
        for (int i = tid; i < 320; i += 256) {
            const int lr = i >> 5, lq = i & 31;
            const int yy = ybase + lr;
            vf4 v = {0.f, 0.f, 0.f, 0.f};
            if (yy >= 0 && yy < H_)
                v = ((const vf4*)(src + yy * W_))[lq];
            float* d = &lds[lr][1 + (lq << 2)];
            d[0] = v.x; d[1] = v.y; d[2] = v.z; d[3] = v.w;
        }
        __syncthreads();

        // compute output row y = pass*8 + r from lds rows r, r+1, r+2
        const int y = pass * 8 + r;
        const float* r0 = &lds[r][x0];               // input row y-1, cols x0-1..x0+4
        const float* r1 = &lds[r + 1][x0];
        const float* r2 = &lds[r + 2][x0];

        float a0 = bv, a1 = bv, a2 = bv, a3 = bv;
        {
            const float p0 = r0[0], p1 = r0[1], p2 = r0[2],
                        p3 = r0[3], p4 = r0[4], p5 = r0[5];
            a0 += w00 * p0 + w01 * p1 + w02 * p2;
            a1 += w00 * p1 + w01 * p2 + w02 * p3;
            a2 += w00 * p2 + w01 * p3 + w02 * p4;
            a3 += w00 * p3 + w01 * p4 + w02 * p5;
        }
        {
            const float p0 = r1[0], p1 = r1[1], p2 = r1[2],
                        p3 = r1[3], p4 = r1[4], p5 = r1[5];
            a0 += w10 * p0 + w11 * p1 + w12 * p2;
            a1 += w10 * p1 + w11 * p2 + w12 * p3;
            a2 += w10 * p2 + w11 * p3 + w12 * p4;
            a3 += w10 * p3 + w11 * p4 + w12 * p5;
        }
        {
            const float p0 = r2[0], p1 = r2[1], p2 = r2[2],
                        p3 = r2[3], p4 = r2[4], p5 = r2[5];
            a0 += w20 * p0 + w21 * p1 + w22 * p2;
            a1 += w20 * p1 + w21 * p2 + w22 * p3;
            a2 += w20 * p2 + w21 * p3 + w22 * p4;
            a3 += w20 * p3 + w21 * p4 + w22 * p5;
        }

        vf4 o = {a0, a1, a2, a3};
        __builtin_nontemporal_store(o, (vf4*)(dst + y * W_) + q);  // b128 store
    }
}

// ---------------------------------------------------------------------------
extern "C" void kernel_launch(void* const* d_in, const int* in_sizes, int n_in,
                              void* d_out, int out_size, void* d_ws, size_t ws_size,
                              hipStream_t stream) {
    const float* x     = (const float*)d_in[0];
    const float* w1    = (const float*)d_in[1];
    const float* gamma = (const float*)d_in[2];
    const float* beta  = (const float*)d_in[3];
    const float* rmean = (const float*)d_in[4];
    const float* rvar  = (const float*)d_in[5];
    const float* w2    = (const float*)d_in[6];
    const float* b2    = (const float*)d_in[7];
    const float* bias  = (const float*)d_in[8];
    float* out = (float*)d_out;

    float* pooled = (float*)d_ws;                    // 32*96   = 3072 floats
    float* wdyn   = pooled + 4096;                   // 32*864  = 27648 floats

    pool_kernel<<<B_ * C_, 256, 0, stream>>>(x, pooled);
    dynw_kernel<<<1, 256, 0, stream>>>(pooled, w1, gamma, beta, rmean, rvar,
                                       w2, b2, wdyn);
    dwconv_kernel<<<B_ * C_, 256, 0, stream>>>(x, wdyn, bias, out);
}
